// GroupedQueryAttention_67542655697063
// MI455X (gfx1250) — matrix operand order repaired and checked by reading, not compiled, not run on hardware
//
#include <hip/hip_runtime.h>
#include <hip/hip_bf16.h>
#include <stdint.h>

// ---------------------------------------------------------------------------
// GroupedQueryAttention for MI455X (gfx1250), wave32 + WMMA f16->f32 +
// ASYNCcnt-tracked global->LDS DMA (global_load_async_to_lds_b128).
//
// Pipeline:
//   convert x -> f16; transpose+convert wq/wk/wv/wo -> f16 [N][K]
//   q/k/v = WMMA GEMMs (f16 out) -> RoPE(q,k) -> transpose V
//   flash attention (f16 tiles, fp32 online softmax) -> ctx f16
//   out = ctx @ wo (WMMA, fp32 out)
// ---------------------------------------------------------------------------

typedef __attribute__((ext_vector_type(16))) _Float16 v16h;
typedef __attribute__((ext_vector_type(8)))  _Float16 v8h;
typedef __attribute__((ext_vector_type(8)))  float    v8f;

#define WMMA_F32_F16(A, B, C) \
  __builtin_amdgcn_wmma_f32_16x16x32_f16(false, (A), false, (B), (short)0, (C), false, false)

// 16-byte global -> LDS async copy (GVS mode: saddr base + 32-bit voffset).
// Tracked by ASYNCcnt; data lands in LDS without touching VGPRs.
__device__ __forceinline__ void async_copy16(uint32_t lds_byte,
                                             const _Float16* base,
                                             uint32_t voff_bytes) {
  asm volatile("global_load_async_to_lds_b128 %0, %1, %2"
               :: "v"(lds_byte), "v"(voff_bytes), "s"(base) : "memory");
}
__device__ __forceinline__ void wait_async0() {
  asm volatile("s_wait_asynccnt 0" ::: "memory");
}
__device__ __forceinline__ uint32_t lds_addr(const void* p) {
  return (uint32_t)(uintptr_t)p;   // low 32 bits of flat shared ptr = LDS offset
}

// ---------------------------------------------------------------------------
// GEMM (TN): C[m][n] = sum_k A[m][k] * Bt[n][k], A/Bt f16, OutT in {f16,f32}.
// BM=BN=128, BK=32. 8 waves, wave = 32x64 sub-tile -> 8 WMMAs per k-step.
// Double-buffered LDS tiles filled by async DMA, prefetch overlaps compute.
// ---------------------------------------------------------------------------
template <typename OutT>
__global__ void __launch_bounds__(256)
gemm_tn_wmma(const _Float16* __restrict__ A, const _Float16* __restrict__ Bt,
             OutT* __restrict__ C, int M, int N, int K)
{
  __shared__ _Float16 As[2][128 * 40];   // [row][k], stride 40 halves (16B align)
  __shared__ _Float16 Bs[2][128 * 40];   // [n][k]

  const int tid  = threadIdx.x;
  const int w    = tid >> 5;
  const int lane = tid & 31;
  const int m    = lane & 15;
  const int h    = lane >> 4;
  const int wrow = (w >> 1) << 5;        // 0,32,64,96
  const int wcol = (w & 1) << 6;         // 0,64
  const int rb   = blockIdx.y * 128;
  const int cb   = blockIdx.x * 128;

  // staging: 512 16B-chunks per tile, 2 per thread per matrix
  const int r0 = tid >> 2;               // rows 0..63  (chunk tid)
  const int r1 = r0 + 64;                // rows 64..127 (chunk tid+256)
  const int c8 = (tid & 3) * 8;          // k sub-offset (halves)

  const _Float16* aB = A  + (size_t)rb * K;
  const _Float16* bB = Bt + (size_t)cb * K;

  auto stage = [&](int buf, int kk) {
    const _Float16* ab = aB + kk;
    const _Float16* bb = bB + kk;
    async_copy16(lds_addr(&As[buf][r0 * 40 + c8]), ab, (uint32_t)((r0 * K + c8) * 2));
    async_copy16(lds_addr(&As[buf][r1 * 40 + c8]), ab, (uint32_t)((r1 * K + c8) * 2));
    async_copy16(lds_addr(&Bs[buf][r0 * 40 + c8]), bb, (uint32_t)((r0 * K + c8) * 2));
    async_copy16(lds_addr(&Bs[buf][r1 * 40 + c8]), bb, (uint32_t)((r1 * K + c8) * 2));
  };

  v8f acc[2][4] = {};
  const int nsteps = K >> 5;
  stage(0, 0);

  for (int s = 0; s < nsteps; ++s) {
    wait_async0();                       // own async stores for tile s done
    __syncthreads();                     // everyone's done; tile s visible,
                                         // buffer s+1 no longer being read
    if (s + 1 < nsteps) stage((s + 1) & 1, (s + 1) << 5);

    const _Float16* Ab = &As[s & 1][0];
    const _Float16* Bb = &Bs[s & 1][0];

    v16h a0, a1, b[4];
    const _Float16* Ar0 = Ab + (wrow + m) * 40;
    const _Float16* Ar1 = Ab + (wrow + 16 + m) * 40;
#pragma unroll
    for (int t = 0; t < 8; ++t) {
      a0[t] = Ar0[h * 8 + t];  a0[8 + t] = Ar0[16 + h * 8 + t];
      a1[t] = Ar1[h * 8 + t];  a1[8 + t] = Ar1[16 + h * 8 + t];
    }
#pragma unroll
    for (int j = 0; j < 4; ++j) {
      const _Float16* Bp = Bb + (wcol + j * 16 + m) * 40;
#pragma unroll
      for (int t = 0; t < 16; ++t) b[j][t] = Bp[h * 8 + t + (t & 8)];
    }
#pragma unroll
    for (int j = 0; j < 4; ++j) {
      acc[0][j] = WMMA_F32_F16(a0, b[j], acc[0][j]);
      acc[1][j] = WMMA_F32_F16(a1, b[j], acc[1][j]);
    }
  }

#pragma unroll
  for (int i = 0; i < 2; ++i)
#pragma unroll
    for (int j = 0; j < 4; ++j)
#pragma unroll
      for (int r = 0; r < 8; ++r) {
        int row = rb + wrow + i * 16 + h * 8 + r;
        int col = cb + wcol + j * 16 + m;
        C[(size_t)row * N + col] = (OutT)acc[i][j][r];
      }
}

// ---------------------------------------------------------------------------
// fp32 -> f16 convert (contiguous)
// ---------------------------------------------------------------------------
__global__ void __launch_bounds__(256)
convert_f32_f16(const float* __restrict__ src, _Float16* __restrict__ dst)
{
  size_t i = (size_t)blockIdx.x * 256 + threadIdx.x;
  dst[i] = (_Float16)src[i];
}

// ---------------------------------------------------------------------------
// W[K][N] fp32 -> Wt[N][K] f16, tiled 32x32 through LDS (coalesced both ways)
// ---------------------------------------------------------------------------
__global__ void __launch_bounds__(256)
transpose_f32_f16(const float* __restrict__ W, _Float16* __restrict__ Wt,
                  int K, int N)
{
  __shared__ float tile[32][33];
  const int tx = threadIdx.x & 31, ty = threadIdx.x >> 5;
  const int n0 = blockIdx.x * 32, k0 = blockIdx.y * 32;
#pragma unroll
  for (int i = 0; i < 4; ++i)
    tile[ty + i * 8][tx] = W[(size_t)(k0 + ty + i * 8) * N + n0 + tx];
  __syncthreads();
#pragma unroll
  for (int i = 0; i < 4; ++i)
    Wt[(size_t)(n0 + ty + i * 8) * K + k0 + tx] = (_Float16)tile[tx][ty + i * 8];
}

// f16 [R][C] -> f16 [C][R]
__global__ void __launch_bounds__(256)
transpose_f16(const _Float16* __restrict__ S, _Float16* __restrict__ D,
              int R, int Cc)
{
  __shared__ _Float16 tile[32][33];
  const int tx = threadIdx.x & 31, ty = threadIdx.x >> 5;
  const int c0 = blockIdx.x * 32, r0 = blockIdx.y * 32;
#pragma unroll
  for (int i = 0; i < 4; ++i)
    tile[ty + i * 8][tx] = S[(size_t)(r0 + ty + i * 8) * Cc + c0 + tx];
  __syncthreads();
#pragma unroll
  for (int i = 0; i < 4; ++i)
    D[(size_t)(c0 + ty + i * 8) * R + r0 + tx] = tile[tx][ty + i * 8];
}

// ---------------------------------------------------------------------------
// RoPE in place on f16 buffer [S][nheads*128], interleaved-pair convention.
// ---------------------------------------------------------------------------
__global__ void __launch_bounds__(256)
rope_f16(_Float16* __restrict__ buf, int nheads)
{
  int idx = blockIdx.x * blockDim.x + threadIdx.x;
  int i  = idx & 63;
  int hh = (idx >> 6) % nheads;
  int s  = idx / (64 * nheads);
  float ang = (float)s * __expf(-(float)i * 0.14391156f);  // ln(1e4)/64
  float sn = __sinf(ang), cs = __cosf(ang);
  _Float16* p = buf + ((size_t)s * nheads + hh) * 128 + 2 * i;
  float x0 = (float)p[0], x1 = (float)p[1];
  p[0] = (_Float16)(x0 * cs - x1 * sn);
  p[1] = (_Float16)(x0 * sn + x1 * cs);
}

// ---------------------------------------------------------------------------
// Flash attention, causal, GQA (32 q heads / 8 kv groups).
// Block = 8 waves = 128 query rows of one head; wave owns 16 rows.
// K tile [32 key][128 c] and pre-transposed V tile [128 c][32 key] staged by
// async DMA into double-buffered LDS; 16 WMMAs per 32-key step.
// ---------------------------------------------------------------------------
__global__ void __launch_bounds__(256)
attn_flash_wmma(const _Float16* __restrict__ q, const _Float16* __restrict__ kh,
                const _Float16* __restrict__ vt, _Float16* __restrict__ ctx)
{
  constexpr int S = 2048, DQ = 4096, DKV = 1024, HD = 128;

  __shared__ _Float16 Ks[2][32 * 136];   // [key][c]
  __shared__ _Float16 Vs[2][128 * 40];   // [c][key]
  __shared__ _Float16 Ps[8][16 * 40];    // per-wave P tile [row][key]

  const int tid  = threadIdx.x;
  const int w    = tid >> 5;
  const int lane = tid & 31;
  const int m    = lane & 15;
  const int h    = lane >> 4;
  const int head = blockIdx.y;
  const int grp  = head >> 2;
  const int qbase = blockIdx.x * 128;
  const int qr    = qbase + w * 16;
  const float scale = 0.08838834764831845f;   // 1/sqrt(128)

  // ---- preload the wave's Q fragments: 2 contiguous 16B loads each ----
  v16h aq[4];
  {
    const _Float16* qp = q + (size_t)(qr + m) * DQ + head * HD;
#pragma unroll
    for (int j = 0; j < 4; ++j) {
      v8h lo = *(const v8h*)(qp + j * 32 + h * 8);
      v8h hi = *(const v8h*)(qp + j * 32 + 16 + h * 8);
#pragma unroll
      for (int t = 0; t < 8; ++t) { aq[j][t] = lo[t]; aq[j][8 + t] = hi[t]; }
    }
  }

  // ---- async staging: 512 chunks each for K and V, 2 per thread ----
  const _Float16* kB = kh + (size_t)grp * HD;
  const _Float16* vB = vt + (size_t)grp * HD * S;
  auto stage = [&](int buf, int kb) {
    const _Float16* kb16 = kB + (size_t)kb * DKV;
    const _Float16* vb16 = vB + kb;
#pragma unroll
    for (int i = 0; i < 2; ++i) {
      int ch = tid + i * 256;
      int kr = ch >> 4, kc = (ch & 15) * 8;
      async_copy16(lds_addr(&Ks[buf][kr * 136 + kc]), kb16,
                   (uint32_t)((kr * DKV + kc) * 2));
      int vr = ch >> 2, vk = (ch & 3) * 8;
      async_copy16(lds_addr(&Vs[buf][vr * 40 + vk]), vb16,
                   (uint32_t)((vr * S + vk) * 2));
    }
  };

  v8f acc[8] = {};
  float mrow[8], lrow[8];
#pragma unroll
  for (int r = 0; r < 8; ++r) { mrow[r] = -3.0e38f; lrow[r] = 0.0f; }

  const int nst = (qbase + 128) >> 5;    // causal: keys < qbase+128
  stage(0, 0);

  for (int s = 0; s < nst; ++s) {
    const int kb = s << 5;
    wait_async0();
    __syncthreads();
    if (s + 1 < nst) stage((s + 1) & 1, kb + 32);
    const _Float16* Kb = &Ks[s & 1][0];
    const _Float16* Vb = &Vs[s & 1][0];

    // ---- scores: Q(16x128) . K^T -> two 16x16 tiles ----
    v8f s0 = {}, s1 = {};
#pragma unroll
    for (int j = 0; j < 4; ++j) {
      v16h b0, b1;
      const _Float16* k0p = Kb + m * 136        + j * 32 + h * 8;
      const _Float16* k1p = Kb + (16 + m) * 136 + j * 32 + h * 8;
#pragma unroll
      for (int t = 0; t < 16; ++t) { b0[t] = k0p[t + (t & 8)]; b1[t] = k1p[t + (t & 8)]; }
      s0 = WMMA_F32_F16(aq[j], b0, s0);
      s1 = WMMA_F32_F16(aq[j], b1, s1);
    }

    // ---- online softmax (fp32), P -> per-wave LDS ----
    _Float16* Pw = &Ps[w][0];
#pragma unroll
    for (int r = 0; r < 8; ++r) {
      int qg = qr + h * 8 + r;
      float x0 = (kb + m      <= qg) ? s0[r] * scale : -3.0e38f;
      float x1 = (kb + 16 + m <= qg) ? s1[r] * scale : -3.0e38f;
      float t = fmaxf(x0, x1);
#pragma unroll
      for (int d = 1; d < 16; d <<= 1) t = fmaxf(t, __shfl_xor(t, d, 32));
      float mn = fmaxf(mrow[r], t);
      float p0 = __expf(x0 - mn);
      float p1 = __expf(x1 - mn);
      float rs = p0 + p1;
#pragma unroll
      for (int d = 1; d < 16; d <<= 1) rs += __shfl_xor(rs, d, 32);
      float corr = __expf(mrow[r] - mn);
      lrow[r] = lrow[r] * corr + rs;
      mrow[r] = mn;
#pragma unroll
      for (int t2 = 0; t2 < 8; ++t2) acc[t2][r] *= corr;
      Pw[(h * 8 + r) * 40 + m]      = (_Float16)p0;
      Pw[(h * 8 + r) * 40 + 16 + m] = (_Float16)p1;
    }
    __builtin_amdgcn_wave_barrier();     // same-wave LDS write->read ordering

    // ---- context: acc += P(16x32) . V(32x128) ----
    v16h ap;
#pragma unroll
    for (int t = 0; t < 8; ++t) {
      ap[t]     = Pw[m * 40 + h * 8 + t];
      ap[8 + t] = Pw[m * 40 + 16 + h * 8 + t];
    }
#pragma unroll
    for (int t = 0; t < 8; ++t) {
      v16h bv;
      const _Float16* vp = Vb + (t * 16 + m) * 40 + h * 8;
#pragma unroll
      for (int i = 0; i < 16; ++i) bv[i] = vp[i + (i & 8)];
      acc[t] = WMMA_F32_F16(ap, bv, acc[t]);
    }
  }

  // ---- normalize, write ctx as f16 [S][H*hd] ----
#pragma unroll
  for (int t = 0; t < 8; ++t)
#pragma unroll
    for (int r = 0; r < 8; ++r) {
      int row = qr + h * 8 + r;
      ctx[(size_t)row * DQ + head * HD + t * 16 + m] =
          (_Float16)(acc[t][r] / lrow[r]);
    }
}

// ---------------------------------------------------------------------------
// Launch sequence.  Workspace (f16): xh 16MB | wqt 32 | wkt 8 | wvt 8 |
// wot 32 | qh 16 | kh 4 | vh 4 | vt 4 | ch 16  = 140 MB
// ---------------------------------------------------------------------------
extern "C" void kernel_launch(void* const* d_in, const int* in_sizes, int n_in,
                              void* d_out, int out_size, void* d_ws, size_t ws_size,
                              hipStream_t stream)
{
  constexpr int S = 2048, D = 4096, KV = 1024;
  const float* x  = (const float*)d_in[0];
  const float* wq = (const float*)d_in[1];
  const float* wk = (const float*)d_in[2];
  const float* wv = (const float*)d_in[3];
  const float* wo = (const float*)d_in[4];
  float* out = (float*)d_out;

  _Float16* p = (_Float16*)d_ws;
  _Float16* xh  = p;  p += (size_t)S * D;
  _Float16* wqt = p;  p += (size_t)D * D;
  _Float16* wkt = p;  p += (size_t)KV * D;
  _Float16* wvt = p;  p += (size_t)KV * D;
  _Float16* wot = p;  p += (size_t)D * D;
  _Float16* qh  = p;  p += (size_t)S * D;
  _Float16* kh  = p;  p += (size_t)S * KV;
  _Float16* vh  = p;  p += (size_t)S * KV;
  _Float16* vt  = p;  p += (size_t)KV * S;
  _Float16* ch  = p;

  dim3 blk(256);

  // precision prep: f16 copies, weights transposed to [N][K]
  convert_f32_f16<<<dim3((S * D) / 256), blk, 0, stream>>>(x, xh);
  transpose_f32_f16<<<dim3(D / 32,  D / 32), blk, 0, stream>>>(wq, wqt, D, D);
  transpose_f32_f16<<<dim3(KV / 32, D / 32), blk, 0, stream>>>(wk, wkt, D, KV);
  transpose_f32_f16<<<dim3(KV / 32, D / 32), blk, 0, stream>>>(wv, wvt, D, KV);
  transpose_f32_f16<<<dim3(D / 32,  D / 32), blk, 0, stream>>>(wo, wot, D, D);

  // projections (f16 out)
  gemm_tn_wmma<_Float16><<<dim3(D / 128,  S / 128), blk, 0, stream>>>(xh, wqt, qh, S, D,  D);
  gemm_tn_wmma<_Float16><<<dim3(KV / 128, S / 128), blk, 0, stream>>>(xh, wkt, kh, S, KV, D);
  gemm_tn_wmma<_Float16><<<dim3(KV / 128, S / 128), blk, 0, stream>>>(xh, wvt, vh, S, KV, D);

  // rope(q), rope(k); V transposed for attention B-fragments
  rope_f16<<<dim3((S * 32 * 64) / 256), blk, 0, stream>>>(qh, 32);
  rope_f16<<<dim3((S * 8  * 64) / 256), blk, 0, stream>>>(kh, 8);
  transpose_f16<<<dim3(KV / 32, S / 32), blk, 0, stream>>>(vh, vt, S, KV);

  attn_flash_wmma<<<dim3(S / 128, 32), blk, 0, stream>>>(qh, kh, vt, ch);

  // output projection (fp32 out)
  gemm_tn_wmma<float><<<dim3(D / 128, S / 128), blk, 0, stream>>>(ch, wot, out, S, D, D);
}
